// HAN_47321949667634
// MI455X (gfx1250) — matrix operand order repaired and checked by reading, not compiled, not run on hardware
//
#include <hip/hip_runtime.h>
#include <cstddef>
#include <cstdint>

#define NE      100000
#define NR      10
#define EDIM    128
#define HID     256
#define NREG    2000
#define NREG_PAD 2016     // NREG rounded up to a multiple of 32 for guard-free WMMA tiles
#define NKG     600000
#define NMP_ENT 40000
#define NMP_REL 5
#define NMP_E   300000
#define BQ      1024
#define ATT     128

typedef __bf16 bf16;
typedef __attribute__((ext_vector_type(16))) __bf16 bf16x16;
typedef __attribute__((ext_vector_type(8)))  float  f32x8;

union FragU { bf16x16 v; uint4 q[2]; };

// ---------------------------------------------------------------------------
// WMMA GEMM (NT): C[M,N] f32 = A[M,K] bf16 (row major) x B[N,K] bf16 (row major:
// the K x N operand pre-transposed so both operands stream contiguous K).
// REQUIRES: M % 32 == 0, N % 32 == 0 (call sites guarantee it; ragged M is padded).
// One wave computes a 32x32 tile: 4 accumulators, A/B fragment reuse, fully
// unrolled K loop (K is a template constant: 128 or 256). Tile indices are
// forced scalar via readfirstlane so no EXEC masking ever touches the WMMAs.
// Epilogue: optional +csrc, +bias[n], activation.
// ---------------------------------------------------------------------------
template <int K>
__global__ __launch_bounds__(256) void gemm_bf16_nt(
    const bf16* __restrict__ A, const bf16* __restrict__ B, float* __restrict__ C,
    int M, int N,
    const float* __restrict__ bias, const float* __restrict__ csrc, int ldsrc,
    int act)
{
    const int tiles_n = N >> 5;
    int wid = blockIdx.x * (blockDim.x >> 5) + (threadIdx.x >> 5);
    wid = __builtin_amdgcn_readfirstlane(wid);          // scalar tile id
    const int tm = wid / tiles_n;
    const int tn = wid - tm * tiles_n;
    const int m0 = tm << 5, n0 = tn << 5;
    if (m0 >= M) return;                                // scalar branch, whole wave exits

    const int lane  = threadIdx.x & 31;
    const int l15   = lane & 15;
    const int half  = lane >> 4;
    const int half8 = half << 3;   // A: K-chunk select per ISA 16-bit A layout
    const int kh16  = half << 4;   // unused: B loads use half8, same K order as A

    f32x8 acc00 = {}, acc01 = {}, acc10 = {}, acc11 = {};

    const bf16* pa0 = A + (size_t)(m0 + l15) * K;
    const bf16* pa1 = pa0 + (size_t)16 * K;
    const bf16* pb0 = B + (size_t)(n0 + l15) * K;
    const bf16* pb1 = pb0 + (size_t)16 * K;

    #pragma unroll
    for (int k = 0; k < K; k += 32) {
        if (k + 32 < K) {   // prefetch next K slab -> global_prefetch
            __builtin_prefetch(pa0 + k + 32, 0, 0);
            __builtin_prefetch(pb0 + k + 32, 0, 0);
        }
        FragU a0, a1, b0, b1;
        // A fragment: lanes 0-15 hold K=[k+0..7]&[k+16..23], lanes 16-31 the +8 chunks
        a0.q[0] = *(const uint4*)(pa0 + k + half8);
        a0.q[1] = *(const uint4*)(pa0 + k + 16 + half8);
        a1.q[0] = *(const uint4*)(pa1 + k + half8);
        a1.q[1] = *(const uint4*)(pa1 + k + 16 + half8);
        // B fragment: same K order as A: lanes 0-15 K=[k+0..7]&[k+16..23], lanes 16-31 the +8 chunks
        b0.q[0] = *(const uint4*)(pb0 + k + half8);
        b0.q[1] = *(const uint4*)(pb0 + k + 16 + half8);
        b1.q[0] = *(const uint4*)(pb1 + k + half8);
        b1.q[1] = *(const uint4*)(pb1 + k + 16 + half8);

        acc00 = __builtin_amdgcn_wmma_f32_16x16x32_bf16(false, a0.v, false, b0.v,
                                                        (short)0, acc00, false, false);
        acc01 = __builtin_amdgcn_wmma_f32_16x16x32_bf16(false, a0.v, false, b1.v,
                                                        (short)0, acc01, false, false);
        acc10 = __builtin_amdgcn_wmma_f32_16x16x32_bf16(false, a1.v, false, b0.v,
                                                        (short)0, acc10, false, false);
        acc11 = __builtin_amdgcn_wmma_f32_16x16x32_bf16(false, a1.v, false, b1.v,
                                                        (short)0, acc11, false, false);
    }

    // C layout: VGPR g -> row mb + half*8 + g, col nb + l15
    auto store_tile = [&](f32x8& acc, int mb, int nb) {
        const int n = nb + l15;
        #pragma unroll
        for (int g = 0; g < 8; ++g) {
            const int m = mb + half * 8 + g;
            float v = acc[g];
            if (csrc) v += csrc[(size_t)m * ldsrc + n];
            if (bias) v += bias[n];
            if (act == 1) v = tanhf(v);
            else if (act == 2) v = fmaxf(v, 0.f);
            C[(size_t)m * N + n] = v;
        }
    };
    store_tile(acc00, m0,      n0);
    store_tile(acc01, m0,      n0 + 16);
    store_tile(acc10, m0 + 16, n0);
    store_tile(acc11, m0 + 16, n0 + 16);
}

// ---------------------------------------------------------------------------
// Support kernels
// ---------------------------------------------------------------------------
__global__ void cvt_f32_to_bf16(const float* __restrict__ in, bf16* __restrict__ out, size_t n) {
    size_t i = (size_t)blockIdx.x * blockDim.x + threadIdx.x;
    if (i < n) out[i] = (bf16)in[i];
}

// in: [batch, K, N] f32  ->  out: [batch, N, K] bf16
__global__ void transpose_cvt(const float* __restrict__ in, bf16* __restrict__ out, int K, int N) {
    const size_t base = (size_t)blockIdx.z * K * N;
    int n = blockIdx.x * blockDim.x + threadIdx.x;
    int k = blockIdx.y * blockDim.y + threadIdx.y;
    if (n < N && k < K)
        out[base + (size_t)n * K + k] = (bf16)in[base + (size_t)k * N + n];
}

__global__ void tanh_cvt(float* __restrict__ x, bf16* __restrict__ xb, size_t n) {
    size_t i = (size_t)blockIdx.x * blockDim.x + threadIdx.x;
    if (i < n) { float v = tanhf(x[i]); x[i] = v; xb[i] = (bf16)v; }
}

__global__ void relu_inplace(float* __restrict__ x, size_t n) {
    size_t i = (size_t)blockIdx.x * blockDim.x + threadIdx.x;
    if (i < n) x[i] = fmaxf(x[i], 0.f);
}

__global__ void copy_f32(const float* __restrict__ in, float* __restrict__ out, size_t n) {
    size_t i = (size_t)blockIdx.x * blockDim.x + threadIdx.x;
    if (i < n) out[i] = in[i];
}

// xp[b,:] = bf16( E[h[b],:] * R[r[b],:] )          block = EDIM threads
__global__ void build_query(const float* __restrict__ E, const float* __restrict__ R,
                            const int* __restrict__ h, const int* __restrict__ r,
                            bf16* __restrict__ xp) {
    int b = blockIdx.x, j = threadIdx.x;
    xp[(size_t)b * EDIM + j] =
        (bf16)(E[(size_t)h[b] * EDIM + j] * R[(size_t)r[b] * EDIM + j]);
}

// dst[i,:] = src[idx[i],:]   (bf16 rows), block = dim threads
__global__ void gather_rows_bf16(const bf16* __restrict__ src, const int* __restrict__ idx,
                                 bf16* __restrict__ dst, int dim) {
    int i = blockIdx.x, j = threadIdx.x;
    dst[(size_t)i * dim + j] = src[(size_t)idx[i] * dim + j];
}

// For edges with etype==rel and dst<maxDst: out[dst,:] += Y[src,:]
__global__ void scatter_add_rel(const float* __restrict__ Y,
                                const int* __restrict__ src, const int* __restrict__ dst,
                                const int* __restrict__ et, int rel, int dim, int maxDst,
                                float* __restrict__ out) {
    int e = blockIdx.x;
    if (et[e] != rel) return;
    int d = dst[e];
    if (d >= maxDst) return;
    int s = src[e];
    const float* yr = Y + (size_t)s * dim;
    float* od = out + (size_t)d * dim;
    for (int j = threadIdx.x; j < dim; j += blockDim.x)
        atomicAdd(&od[j], yr[j]);
}

// wout[reg*2+p] = ( tanh(z[reg,:] @ w1 + b1) @ w2 );  block = ATT threads
__global__ void att_proj(const float* __restrict__ z, const float* __restrict__ w1,
                         const float* __restrict__ b1, const float* __restrict__ w2,
                         float* __restrict__ wout, int p) {
    __shared__ float red[ATT];
    int reg = blockIdx.x, a = threadIdx.x;
    const float* zr = z + (size_t)reg * HID;
    float s = 0.f;
    for (int h = 0; h < HID; ++h) s += zr[h] * w1[(size_t)h * ATT + a];
    red[a] = tanhf(s + b1[a]) * w2[a];
    __syncthreads();
    for (int off = ATT / 2; off > 0; off >>= 1) {
        if (a < off) red[a] += red[a + off];
        __syncthreads();
    }
    if (a == 0) wout[reg * 2 + p] = red[0];
}

// beta = softmax over the two metapath means
__global__ void beta_kernel(const float* __restrict__ wout, float* __restrict__ beta) {
    __shared__ float s0[256], s1[256];
    int t = threadIdx.x;
    float a = 0.f, b = 0.f;
    for (int i = t; i < NREG; i += 256) { a += wout[i * 2]; b += wout[i * 2 + 1]; }
    s0[t] = a; s1[t] = b; __syncthreads();
    for (int off = 128; off > 0; off >>= 1) {
        if (t < off) { s0[t] += s0[t + off]; s1[t] += s1[t + off]; }
        __syncthreads();
    }
    if (t == 0) {
        float m0 = s0[0] / (float)NREG, m1 = s1[0] / (float)NREG;
        float mx = fmaxf(m0, m1);
        float e0 = __expf(m0 - mx), e1 = __expf(m1 - mx);
        float inv = 1.f / (e0 + e1);
        beta[0] = e0 * inv; beta[1] = e1 * inv;
    }
}

__global__ void combine(const float* __restrict__ s0, const float* __restrict__ s1,
                        const float* __restrict__ beta, bf16* __restrict__ hb, size_t n) {
    size_t i = (size_t)blockIdx.x * blockDim.x + threadIdx.x;
    if (i < n) hb[i] = (bf16)(beta[0] * s0[i] + beta[1] * s1[i]);
}

// ---------------------------------------------------------------------------
static void launch_gemm(const bf16* A, const bf16* B, float* C, int M, int N, int K,
                        const float* bias, const float* csrc, int ldsrc, int act,
                        hipStream_t s) {
    int tiles = (M / 32) * (N / 32);
    int blocks = (tiles + 7) / 8;   // 8 waves (wave32) per 256-thread block
    if (K == 128)
        gemm_bf16_nt<128><<<blocks, 256, 0, s>>>(A, B, C, M, N, bias, csrc, ldsrc, act);
    else
        gemm_bf16_nt<256><<<blocks, 256, 0, s>>>(A, B, C, M, N, bias, csrc, ldsrc, act);
}

extern "C" void kernel_launch(void* const* d_in, const int* in_sizes, int n_in,
                              void* d_out, int out_size, void* d_ws, size_t ws_size,
                              hipStream_t stream) {
    const float* E_weight   = (const float*)d_in[0];
    const float* R_weight   = (const float*)d_in[1];
    const float* rgcn_W     = (const float*)d_in[2];
    const float* rgcn_Wroot = (const float*)d_in[3];
    const float* rgcn_b     = (const float*)d_in[4];
    const float* mpW[2]     = {(const float*)d_in[5], (const float*)d_in[8]};
    const float* mpWroot[2] = {(const float*)d_in[6], (const float*)d_in[9]};
    const float* mpb[2]     = {(const float*)d_in[7], (const float*)d_in[10]};
    const float* sa_w1      = (const float*)d_in[11];
    const float* sa_b1      = (const float*)d_in[12];
    const float* sa_w2      = (const float*)d_in[13];
    const float* pred_W     = (const float*)d_in[14];
    const float* pred_b     = (const float*)d_in[15];
    const int*   h_idx      = (const int*)d_in[16];
    const int*   r_idx      = (const int*)d_in[17];
    const int*   kg_ei      = (const int*)d_in[18];
    const int*   kg_et      = (const int*)d_in[19];
    const int*   mp_ei[2]   = {(const int*)d_in[20], (const int*)d_in[23]};
    const int*   mp_et[2]   = {(const int*)d_in[21], (const int*)d_in[24]};
    const int*   mp_eids[2] = {(const int*)d_in[22], (const int*)d_in[25]};

    float* out_Ereg = (float*)d_out;                               // [NREG, EDIM]
    float* out_pred = (float*)d_out + (size_t)NREG * EDIM;         // [BQ, NE]

    // ---- workspace carve-out (256B aligned slabs) ----
    size_t off = 0;
    char* wsb = (char*)d_ws;
    auto alloc = [&](size_t bytes) -> void* {
        void* p = wsb + off;
        off += (bytes + 255) & ~(size_t)255;
        return p;
    };
    bf16*  Xb        = (bf16*)alloc((size_t)NE * EDIM * 2);
    bf16*  rgcnWt    = (bf16*)alloc((size_t)NR * EDIM * EDIM * 2);
    bf16*  rootWt    = (bf16*)alloc((size_t)EDIM * EDIM * 2);
    bf16*  mpWtB     = (bf16*)alloc((size_t)NMP_REL * EDIM * HID * 2);
    bf16*  mpRootWtB = (bf16*)alloc((size_t)EDIM * HID * 2);
    bf16*  predWt    = (bf16*)alloc((size_t)HID * EDIM * 2);
    float* Eout      = (float*)alloc((size_t)NE * EDIM * 4);
    bf16*  Eb        = (bf16*)alloc((size_t)NE * EDIM * 2);
    float* Yr        = (float*)alloc((size_t)NE * EDIM * 4);
    bf16*  Xp        = (bf16*)alloc((size_t)BQ * EDIM * 2);
    bf16*  Xsub      = (bf16*)alloc((size_t)NMP_ENT * EDIM * 2);
    float* Ymp       = (float*)alloc((size_t)NMP_ENT * HID * 4);
    float* sem[2]    = {(float*)alloc((size_t)NREG_PAD * HID * 4),    // padded M
                        (float*)alloc((size_t)NREG_PAD * HID * 4)};
    float* wsc       = (float*)alloc((size_t)NREG * 2 * 4);
    float* betaP     = (float*)alloc(2 * 4);
    bf16*  hb        = (bf16*)alloc((size_t)NREG_PAD * HID * 2);      // padded M
    float* EregPad   = (float*)alloc((size_t)NREG_PAD * EDIM * 4);    // padded C

    const size_t nE = (size_t)NE * EDIM;
    dim3 tb(16, 16);

    // 1) bf16 copy of entity table
    cvt_f32_to_bf16<<<(int)((nE + 255) / 256), 256, 0, stream>>>(E_weight, Xb, nE);

    // 2) weight transposes (K x N -> N x K bf16) for contiguous-K WMMA operands
    transpose_cvt<<<dim3(EDIM / 16, EDIM / 16, NR), tb, 0, stream>>>(rgcn_W, rgcnWt, EDIM, EDIM);
    transpose_cvt<<<dim3(EDIM / 16, EDIM / 16, 1),  tb, 0, stream>>>(rgcn_Wroot, rootWt, EDIM, EDIM);
    transpose_cvt<<<dim3(EDIM / 16, HID / 16, 1),   tb, 0, stream>>>(pred_W, predWt, HID, EDIM);

    // 3) main RGCN: root term, then transform-and-scatter per relation
    launch_gemm(Xb, rootWt, Eout, NE, EDIM, EDIM, rgcn_b, nullptr, 0, 0, stream);
    for (int r = 0; r < NR; ++r) {
        launch_gemm(Xb, rgcnWt + (size_t)r * EDIM * EDIM, Yr, NE, EDIM, EDIM,
                    nullptr, nullptr, 0, 0, stream);
        scatter_add_rel<<<NKG, 128, 0, stream>>>(Yr, kg_ei, kg_ei + NKG, kg_et,
                                                 r, EDIM, NE, Eout);
    }
    // 4) E = tanh(...), keep bf16 copy for the scoring GEMM
    tanh_cvt<<<(int)((nE + 255) / 256), 256, 0, stream>>>(Eout, Eb, nE);

    // 5) KGE scoring: pred = (E[h] * R[r]) @ E^T   (NT GEMM, B = Eb rows = K-contig)
    build_query<<<BQ, EDIM, 0, stream>>>(Eout, R_weight, h_idx, r_idx, Xp);
    launch_gemm(Xp, Eb, out_pred, BQ, NE, EDIM, nullptr, nullptr, 0, 0, stream);

    // 6) metapath RGCNs (only rows < NREG survive the slice; GEMMs padded to NREG_PAD)
    for (int p = 0; p < 2; ++p) {
        gather_rows_bf16<<<NMP_ENT, EDIM, 0, stream>>>(Eb, mp_eids[p], Xsub, EDIM);
        transpose_cvt<<<dim3(HID / 16, EDIM / 16, NMP_REL), tb, 0, stream>>>(mpW[p], mpWtB, EDIM, HID);
        transpose_cvt<<<dim3(HID / 16, EDIM / 16, 1),       tb, 0, stream>>>(mpWroot[p], mpRootWtB, EDIM, HID);
        launch_gemm(Xsub, mpRootWtB, sem[p], NREG_PAD, HID, EDIM, mpb[p], nullptr, 0, 0, stream);
        for (int r = 0; r < NMP_REL; ++r) {
            launch_gemm(Xsub, mpWtB + (size_t)r * EDIM * HID, Ymp, NMP_ENT, HID, EDIM,
                        nullptr, nullptr, 0, 0, stream);
            scatter_add_rel<<<NMP_E, 256, 0, stream>>>(Ymp, mp_ei[p], mp_ei[p] + NMP_E,
                                                       mp_et[p], r, HID, NREG, sem[p]);
        }
        relu_inplace<<<(int)(((size_t)NREG * HID + 255) / 256), 256, 0, stream>>>(
            sem[p], (size_t)NREG * HID);
        att_proj<<<NREG, ATT, 0, stream>>>(sem[p], sa_w1, sa_b1, sa_w2, wsc, p);
    }

    // 7) semantic attention combine (fill padded hb rows too; pad rows feed pad C rows only)
    beta_kernel<<<1, 256, 0, stream>>>(wsc, betaP);
    combine<<<(int)(((size_t)NREG_PAD * HID + 255) / 256), 256, 0, stream>>>(
        sem[0], sem[1], betaP, hb, (size_t)NREG_PAD * HID);

    // 8) E_reg = h @ pred_W + pred_b + E[:NREG]  (padded GEMM into ws, then copy 2000 rows)
    launch_gemm(hb, predWt, EregPad, NREG_PAD, EDIM, HID, pred_b, Eout, EDIM, 0, stream);
    copy_f32<<<(int)(((size_t)NREG * EDIM + 255) / 256), 256, 0, stream>>>(
        EregPad, out_Ereg, (size_t)NREG * EDIM);
}
